// TensorGNAN_61022895341784
// MI455X (gfx1250) — matrix operand order repaired and checked by reading, not compiled, not run on hardware
//
#include <hip/hip_runtime.h>
#include <hip/hip_bf16.h>

typedef __attribute__((ext_vector_type(16))) _Float16 v16h;
typedef __attribute__((ext_vector_type(2)))  _Float16 h2;
typedef __attribute__((ext_vector_type(8)))  float    v8f;

namespace {
constexpr int kN = 3072;   // nodes
constexpr int kF = 128;    // in features
constexpr int kH = 16;     // hidden
constexpr int kO = 8;      // out channels
constexpr int kNTP = kN / 32;         // 96 n-tile PAIRS (32 rows each)
constexpr int kMSPLIT = 32;           // m-range splits
constexpr int kMCHUNK = kN / kMSPLIT; // 96 m per wave
}

// ---------------------------------------------------------------------------
// Phase 1: g[n,o] = sum_f ( relu(x[n,f]*W1[f,:]+b1[f,:]) @ W2[f,:,:] + b2[f,:] )
// ---------------------------------------------------------------------------
__global__ __launch_bounds__(256) void gnan_g(
    const float* __restrict__ x,  const float* __restrict__ W1,
    const float* __restrict__ b1, const float* __restrict__ W2,
    const float* __restrict__ b2, float* __restrict__ g)
{
  int n = blockIdx.x * blockDim.x + threadIdx.x;
  if (n >= kN) return;
  float acc[kO];
#pragma unroll
  for (int o = 0; o < kO; ++o) acc[o] = 0.f;
  for (int f = 0; f < kF; ++f) {
    float xv = x[n * kF + f];
#pragma unroll
    for (int o = 0; o < kO; ++o) acc[o] += b2[f * kO + o];
#pragma unroll
    for (int h = 0; h < kH; ++h) {
      float hv = fmaxf(fmaf(xv, W1[f * kH + h], b1[f * kH + h]), 0.f);
      const float* w2 = W2 + (size_t)(f * kH + h) * kO;
#pragma unroll
      for (int o = 0; o < kO; ++o) acc[o] = fmaf(hv, w2[o], acc[o]);
    }
  }
#pragma unroll
  for (int o = 0; o < kO; ++o) g[n * kO + o] = acc[o];
}

// ---------------------------------------------------------------------------
// Gsum[o] = sum_m g[m,o]   (one block per o)
// ---------------------------------------------------------------------------
__global__ __launch_bounds__(256) void gnan_gsum(
    const float* __restrict__ g, float* __restrict__ Gsum)
{
  __shared__ float s[256];
  int o = blockIdx.x;
  float a = 0.f;
  for (int m = threadIdx.x; m < kN; m += 256) a += g[m * kO + o];
  s[threadIdx.x] = a;
  __syncthreads();
  for (int st = 128; st > 0; st >>= 1) {
    if ((int)threadIdx.x < st) s[threadIdx.x] += s[threadIdx.x + st];
    __syncthreads();
  }
  if (threadIdx.x == 0) Gsum[o] = s[0];
}

// ---------------------------------------------------------------------------
// Vpad[p*32+l][s] = V[(2p+kk/16)*16+kk%16, n=l%16], kk = table-order k(l/16,s); V[m*16+h,o] = (o<8) ? rW2[h,o]*g[m,o] : 0
// ---------------------------------------------------------------------------
__global__ __launch_bounds__(256) void gnan_vpad(
    const float* __restrict__ g, const float* __restrict__ rW2,
    v16h* __restrict__ Vpad)
{
  int k = blockIdx.x * blockDim.x + threadIdx.x;
  if (k >= kN * kH) return;
  int p = k >> 5;
  int o = k & 15;
  int hh = (k >> 4) & 1;
  v16h v;
#pragma unroll
  for (int s = 0; s < 16; ++s) {
    int kk = (s < 8) ? (8 * hh + s) : (16 + 8 * hh + (s - 8));
    int m = 2 * p + (kk >> 4);
    int h = kk & 15;
    v[s] = (_Float16)((o < kO) ? rW2[h * kO + o] * g[m * kO + o] : 0.f);
  }
  Vpad[k] = v;
}

// ---------------------------------------------------------------------------
// out[n,o] = rb2[o] * Gsum[o]   (bias term; GEMM atomically accumulates on top)
// ---------------------------------------------------------------------------
__global__ __launch_bounds__(256) void gnan_outinit(
    const float* __restrict__ rb2, const float* __restrict__ Gsum,
    float* __restrict__ out)
{
  int i = blockIdx.x * blockDim.x + threadIdx.x;
  if (i < kN * kO) {
    int o = i % kO;
    out[i] = rb2[o] * Gsum[o];
  }
}

// ---------------------------------------------------------------------------
// Main fused GEMM: out[n,o] += sum_{m,h} relu(d[n,m]*rW1[h]+rb1[h]) * V[m*16+h,o]
// One wave per (32-row n-tile-pair, 96-m chunk). Per 4-m body: float4 d loads,
// 2 B rows (v16h) reused across both n-tiles -> 4 WMMAs.
// ---------------------------------------------------------------------------
union AFrag { v16h v; h2 p[8]; };

__device__ __forceinline__ v16h make_a(float d0, float d1,
                                       const h2* w1h, const h2* b1h)
{
  _Float16 dh0 = (_Float16)d0;
  _Float16 dh1 = (_Float16)d1;
  h2 db0 = {dh0, dh0};
  h2 db1 = {dh1, dh1};
  h2 zero = {(_Float16)0.f, (_Float16)0.f};
  AFrag a;
#pragma unroll
  for (int t = 0; t < 4; ++t)
    a.p[t] = __builtin_elementwise_max(db0 * w1h[t] + b1h[t], zero);
#pragma unroll
  for (int t = 0; t < 4; ++t)
    a.p[4 + t] = __builtin_elementwise_max(db1 * w1h[t] + b1h[t], zero);
  return a.v;
}

__global__ __launch_bounds__(256) void gnan_main(
    const float* __restrict__ dist, const float* __restrict__ norm,
    const float* __restrict__ rW1,  const float* __restrict__ rb1,
    const v16h* __restrict__ Vpad,  float* __restrict__ out)
{
  const int lane = threadIdx.x & 31;
  const int wid  = (blockIdx.x * blockDim.x + threadIdx.x) >> 5;
  const int ntp  = wid % kNTP;   // n-tile pair: rows [ntp*32, ntp*32+32)
  const int mspl = wid / kNTP;

  const int row0  = ntp * 32 + (lane & 15);  // tile 0 row for this lane
  const int hbase = (lane < 16) ? 0 : 8;     // A-fragment h-range per lane half

  // loop-invariant rho weights, packed f16 pairs
  h2 w1h[4], b1h[4];
#pragma unroll
  for (int t = 0; t < 4; ++t) {
    w1h[t] = h2{(_Float16)rW1[hbase + 2 * t], (_Float16)rW1[hbase + 2 * t + 1]};
    b1h[t] = h2{(_Float16)rb1[hbase + 2 * t], (_Float16)rb1[hbase + 2 * t + 1]};
  }

  const float* dr0 = dist + (size_t)row0 * kN;
  const float* nr0 = norm + (size_t)row0 * kN;
  const float* dr1 = dr0 + (size_t)16 * kN;
  const float* nr1 = nr0 + (size_t)16 * kN;

  v8f c0 = {};
  v8f c1 = {};
  const int m0 = mspl * kMCHUNK;
  for (int mm = m0; mm < m0 + kMCHUNK; mm += 4) {
    // B rows for K-chunks [mm,mm+2) and [mm+2,mm+4): row = mm*16 + K_local
    v16h bA = Vpad[(size_t)mm * 16 + lane];
    v16h bB = Vpad[(size_t)(mm + 2) * 16 + lane];

    float4 dA = *(const float4*)(dr0 + mm);
    float4 nA = *(const float4*)(nr0 + mm);
    float4 dB = *(const float4*)(dr1 + mm);
    float4 nB = *(const float4*)(nr1 + mm);

    float a0 = dA.x * __builtin_amdgcn_rcpf(nA.x);
    float a1 = dA.y * __builtin_amdgcn_rcpf(nA.y);
    float a2 = dA.z * __builtin_amdgcn_rcpf(nA.z);
    float a3 = dA.w * __builtin_amdgcn_rcpf(nA.w);
    float e0 = dB.x * __builtin_amdgcn_rcpf(nB.x);
    float e1 = dB.y * __builtin_amdgcn_rcpf(nB.y);
    float e2 = dB.z * __builtin_amdgcn_rcpf(nB.z);
    float e3 = dB.w * __builtin_amdgcn_rcpf(nB.w);

    v16h a00 = make_a(a0, a1, w1h, b1h);   // tile0, m {mm, mm+1}
    v16h a01 = make_a(a2, a3, w1h, b1h);   // tile0, m {mm+2, mm+3}
    v16h a10 = make_a(e0, e1, w1h, b1h);   // tile1, m {mm, mm+1}
    v16h a11 = make_a(e2, e3, w1h, b1h);   // tile1, m {mm+2, mm+3}

    c0 = __builtin_amdgcn_wmma_f32_16x16x32_f16(false, a00, false, bA, (short)0, c0, false, false);
    c1 = __builtin_amdgcn_wmma_f32_16x16x32_f16(false, a10, false, bA, (short)0, c1, false, false);
    c0 = __builtin_amdgcn_wmma_f32_16x16x32_f16(false, a01, false, bB, (short)0, c0, false, false);
    c1 = __builtin_amdgcn_wmma_f32_16x16x32_f16(false, a11, false, bB, (short)0, c1, false, false);
  }

  // C fragment: VGPR j -> M = j + 8*(lane/16), N = lane%16
  const int o = lane & 15;
  if (o < kO) {
    const int rb0 = ntp * 32 + ((lane < 16) ? 0 : 8);
#pragma unroll
    for (int j = 0; j < 8; ++j) {
      atomicAdd(out + (size_t)(rb0 + j) * kO + o, c0[j]);
      atomicAdd(out + (size_t)(rb0 + 16 + j) * kO + o, c1[j]);
    }
  }
}

extern "C" void kernel_launch(void* const* d_in, const int* in_sizes, int n_in,
                              void* d_out, int out_size, void* d_ws, size_t ws_size,
                              hipStream_t stream) {
  const float* x    = (const float*)d_in[0];
  // d_in[1] = edge_index (int64) — unused by the reference computation
  const float* dist = (const float*)d_in[2];
  const float* norm = (const float*)d_in[3];
  const float* W1   = (const float*)d_in[4];
  const float* b1   = (const float*)d_in[5];
  const float* W2   = (const float*)d_in[6];
  const float* b2   = (const float*)d_in[7];
  const float* rW1  = (const float*)d_in[8];
  const float* rb1  = (const float*)d_in[9];
  const float* rW2  = (const float*)d_in[10];
  const float* rb2  = (const float*)d_in[11];
  float* out = (float*)d_out;

  // workspace layout
  char* ws = (char*)d_ws;
  float* g    = (float*)ws;                    // kN*kO f32      = 98304 B
  float* Gsum = (float*)(ws + 98304);          // 8 f32
  v16h*  Vpad = (v16h*)(ws + 98432);           // kN*kH rows * 32 B = 1.5 MB

  gnan_g      <<<kN / 256,          256, 0, stream>>>(x, W1, b1, W2, b2, g);
  gnan_gsum   <<<kO,                256, 0, stream>>>(g, Gsum);
  gnan_vpad   <<<(kN * kH) / 256,   256, 0, stream>>>(g, rW2, Vpad);
  gnan_outinit<<<(kN * kO) / 256,   256, 0, stream>>>(rb2, Gsum, out);

  // 96 tile-pairs * 32 m-splits = 3072 waves = 384 blocks of 8 waves
  gnan_main<<<(kNTP * kMSPLIT) / 8, 256, 0, stream>>>(dist, norm, rW1, rb1, Vpad, out);
}
